// RnntBeamSearchModule_9380208574595
// MI455X (gfx1250) — matrix operand order repaired and checked by reading, not compiled, not run on hardware
//
#include <hip/hip_runtime.h>

#define BB     128
#define TT     160
#define ENCD   1024
#define EMBD   512
#define PREDD  640
#define JOINTD 512
#define VV     1024
#define GATES  2560
#define ITERS  48

typedef __bf16 bf16_t;
typedef __attribute__((ext_vector_type(8)))  __bf16 v8bf;
typedef __attribute__((ext_vector_type(16))) __bf16 v16bf;
typedef __attribute__((ext_vector_type(8)))  float  v8f;

__device__ __forceinline__ float sigmf(float x) { return 1.0f / (1.0f + expf(-x)); }

// One wave computes a 16x64 f32 tile (4 adjacent 16x16 WMMA tiles) of
// X[M,K](row-major bf16) @ Wt[N,K](bf16, i.e. W^T). A fragment loaded once per
// K-step, reused across 4 B fragments. Two physical fragment sets alternate by
// K-step parity (no inter-set copies -> allocator cannot coalesce them), so the
// loads for step s+1 are in flight while the WMMAs of step s execute.
// A lane layout (16-bit A 16x32): lane<16 -> row m0+r, K {kk..kk+7, kk+16..kk+23};
//                                 lane>=16 -> same row, K {kk+8..15, kk+24..31}.
// B lane layout: lane<16 -> col n+r, K {kk..kk+7, kk+16..kk+23}; lane>=16 -> col n+r, K {kk+8..15, kk+24..31}.
template <int K>
__device__ __forceinline__ void wmma_accum4(const bf16_t* __restrict__ A, int lda,
                                            const bf16_t* __restrict__ Bt, int ldb,
                                            int m0, int n0,
                                            const int* __restrict__ rowIdx, v8f acc[4]) {
  const int lane = threadIdx.x & 31;
  const int r    = lane & 15;
  const int half = lane >> 4;
  const int row  = rowIdx ? rowIdx[m0 + r] : (m0 + r);
  const bf16_t* arow = A  + (long)row * lda;
  const bf16_t* bc   = Bt + (long)(n0 + r) * ldb;

  auto ldA = [&](int kk, v16bf& a) {
    v8bf a0 = *(const v8bf*)(arow + kk + half * 8);
    v8bf a1 = *(const v8bf*)(arow + kk + 16 + half * 8);
#pragma unroll
    for (int i = 0; i < 8; ++i) { a[i] = a0[i]; a[8 + i] = a1[i]; }
  };
  auto ldB = [&](int kk, int nt, v16bf& b) {
    const bf16_t* p = bc + (long)(nt * 16) * ldb + kk + half * 8;
    v8bf b0 = *(const v8bf*)(p);
    v8bf b1 = *(const v8bf*)(p + 16);
#pragma unroll
    for (int i = 0; i < 8; ++i) { b[i] = b0[i]; b[8 + i] = b1[i]; }
  };

  v16bf aA, aB;
  v16bf bA[4], bB[4];
  ldA(0, aA);
#pragma unroll
  for (int nt = 0; nt < 4; ++nt) ldB(0, nt, bA[nt]);

  constexpr int NS = K / 32;
#pragma unroll
  for (int s = 0; s < NS; ++s) {
    const int kk = s * 32;
    if (s + 1 < NS) {                       // prefetch next step into the other set
      if ((s & 1) == 0) {
        ldA(kk + 32, aB);
#pragma unroll
        for (int nt = 0; nt < 4; ++nt) ldB(kk + 32, nt, bB[nt]);
      } else {
        ldA(kk + 32, aA);
#pragma unroll
        for (int nt = 0; nt < 4; ++nt) ldB(kk + 32, nt, bA[nt]);
      }
    }
    if ((s & 1) == 0) {
#pragma unroll
      for (int nt = 0; nt < 4; ++nt)
        acc[nt] = __builtin_amdgcn_wmma_f32_16x16x32_bf16(false, aA, false, bA[nt],
                                                          (short)0, acc[nt], false, false);
    } else {
#pragma unroll
      for (int nt = 0; nt < 4; ++nt)
        acc[nt] = __builtin_amdgcn_wmma_f32_16x16x32_bf16(false, aB, false, bB[nt],
                                                          (short)0, acc[nt], false, false);
    }
  }
}

// ---- one-time weight prep ----
__global__ void transpose_cast_kernel(const float* __restrict__ W, bf16_t* __restrict__ Wt,
                                      int K, int N) {
  long idx = (long)blockIdx.x * 256 + threadIdx.x;
  if (idx >= (long)K * N) return;
  int k = (int)(idx / N), n = (int)(idx % N);
  Wt[(long)n * K + k] = (bf16_t)W[idx];
}

__global__ void cast_kernel(const float* __restrict__ X, bf16_t* __restrict__ Y, long n) {
  long idx = (long)blockIdx.x * 256 + threadIdx.x;
  if (idx < n) Y[idx] = (bf16_t)X[idx];
}

// ---- init: g0 = E[0] @ Wi + bi + bh (h0 = 0) ----
__global__ void init_gates_kernel(const float* __restrict__ E, const float* __restrict__ Wi,
                                  const float* __restrict__ bi, const float* __restrict__ bh,
                                  float* __restrict__ g0) {
  int j = blockIdx.x * 256 + threadIdx.x;
  if (j >= GATES) return;
  float s = bi[j] + bh[j];
  for (int k = 0; k < EMBD; ++k) s += E[k] * Wi[(long)k * GATES + j];
  g0[j] = s;
}

__global__ void init_state_kernel(const float* __restrict__ g0, float* __restrict__ h,
                                  float* __restrict__ c, int* enc_t, int* n_em,
                                  float* scores, int* active) {
  int b = blockIdx.x;
  int j = blockIdx.y * 256 + threadIdx.x;
  if (blockIdx.y == 0 && threadIdx.x == 0) { enc_t[b] = 0; n_em[b] = 0; scores[b] = 0.f; active[b] = 1; }
  if (j >= PREDD) return;
  float c0 = sigmf(g0[j]) * tanhf(g0[1280 + j]);        // f*0 + i*tanh(g)
  float h0 = tanhf(c0) * sigmf(g0[1920 + j]);
  c[(long)b * PREDD + j] = c0;
  h[(long)b * PREDD + j] = h0;
}

// ---- per-step: gather encoder frame + cast h (pred_out == h) to bf16 ----
__global__ void prep_kernel(const float* __restrict__ enc_out, const int* __restrict__ enc_t,
                            const float* __restrict__ h, bf16_t* __restrict__ enc_bf,
                            bf16_t* __restrict__ h_bf) {
  int b = blockIdx.x;
  int j = blockIdx.y * 256 + threadIdx.x;          // j < 1024
  int t = enc_t[b]; t = t < 0 ? 0 : (t > TT - 1 ? TT - 1 : t);
  enc_bf[(long)b * ENCD + j] = (bf16_t)enc_out[((long)b * TT + t) * ENCD + j];
  if (j < PREDD) h_bf[(long)b * PREDD + j] = (bf16_t)h[(long)b * PREDD + j];
}

// ---- joint = tanh(enc @ W_enc + h @ W_pred + b_joint), stored bf16 ----
__global__ void __launch_bounds__(32, 1)
joint_kernel(const bf16_t* __restrict__ encA, const bf16_t* __restrict__ predA,
             const bf16_t* __restrict__ WencT, const bf16_t* __restrict__ WpredT,
             const float* __restrict__ bj, bf16_t* __restrict__ Jbf) {
  int tile = blockIdx.x;                 // 8 Mtiles * (512/64=8) Ngroups
  int m0 = (tile >> 3) * 16, n0 = (tile & 7) * 64;
  v8f acc[4] = {};
  wmma_accum4<ENCD>(encA, ENCD, WencT, ENCD, m0, n0, nullptr, acc);
  wmma_accum4<PREDD>(predA, PREDD, WpredT, PREDD, m0, n0, nullptr, acc);
  int lane = threadIdx.x & 31, r = lane & 15, half = lane >> 4;
#pragma unroll
  for (int nt = 0; nt < 4; ++nt) {
    int n = n0 + nt * 16 + r;
    float bjv = bj[n];
#pragma unroll
    for (int i = 0; i < 8; ++i)
      Jbf[(long)(m0 + half * 8 + i) * JOINTD + n] = (bf16_t)tanhf(acc[nt][i] + bjv);
  }
}

// ---- logits = joint @ W_out ----
__global__ void __launch_bounds__(32, 1)
logits_kernel(const bf16_t* __restrict__ Jbf, const bf16_t* __restrict__ WoutT,
              float* __restrict__ logits) {
  int tile = blockIdx.x;                 // 8 * (1024/64=16)
  int m0 = (tile >> 4) * 16, n0 = (tile & 15) * 64;
  v8f acc[4] = {};
  wmma_accum4<JOINTD>(Jbf, JOINTD, WoutT, JOINTD, m0, n0, nullptr, acc);
  int lane = threadIdx.x & 31, r = lane & 15, half = lane >> 4;
#pragma unroll
  for (int nt = 0; nt < 4; ++nt) {
    int n = n0 + nt * 16 + r;
#pragma unroll
    for (int i = 0; i < 8; ++i)
      logits[(long)(m0 + half * 8 + i) * VV + n] = acc[nt][i];
  }
}

// ---- per-row log-softmax + extended-vocab argmax + decode state update ----
__global__ void decide_kernel(const float* __restrict__ logits, const int* __restrict__ enc_len,
                              int* enc_t, int* n_em, float* scores, int* active,
                              int* tok_sel, int* upd, float* __restrict__ out, int step) {
  int b = blockIdx.x, tid = threadIdx.x;
  const float* row = logits + (long)b * VV;
  __shared__ float red[256];
  __shared__ int   redi[256];
  // row max
  float mx = -INFINITY;
  for (int j = tid; j < VV; j += 256) mx = fmaxf(mx, row[j]);
  red[tid] = mx; __syncthreads();
  for (int s = 128; s > 0; s >>= 1) { if (tid < s) red[tid] = fmaxf(red[tid], red[tid + s]); __syncthreads(); }
  float rowmax = red[0]; __syncthreads();
  // logsumexp
  float sm = 0.f;
  for (int j = tid; j < VV; j += 256) sm += expf(row[j] - rowmax);
  red[tid] = sm; __syncthreads();
  for (int s = 128; s > 0; s >>= 1) { if (tid < s) red[tid] += red[tid + s]; __syncthreads(); }
  float lse = rowmax + logf(red[0]); __syncthreads();
  // argmax over non-blank columns 1..V-1 (shift-invariant vs log_softmax), first-index ties
  float bv = -INFINITY; int bi_ = VV + 1;
  for (int j = tid; j < VV; j += 256)
    if (j >= 1) { float v = row[j]; if (v > bv) { bv = v; bi_ = j; } }
  red[tid] = bv; redi[tid] = bi_; __syncthreads();
  for (int s = 128; s > 0; s >>= 1) {
    if (tid < s) {
      float v2 = red[tid + s]; int i2 = redi[tid + s];
      if (v2 > red[tid] || (v2 == red[tid] && i2 < redi[tid])) { red[tid] = v2; redi[tid] = i2; }
    }
    __syncthreads();
  }
  if (tid == 0) {
    float blankv = row[0];                      // ext column V is copy of original blank col
    int   oa = active[b];
    bool  force = n_em[b] >= 2;                 // step_max_symbols
    int tok; float tval;
    if (force) { tok = VV; tval = blankv; }
    else if (red[0] >= blankv) { tok = redi[0]; tval = red[0]; }   // non-blank wins ties (lower index)
    else { tok = VV; tval = blankv; }
    float tlp = tval - lse;
    bool is_blank = (tok == VV);
    if (oa) scores[b] += tlp;
    upd[b] = (oa && !is_blank) ? 1 : 0;
    int et = enc_t[b] + ((oa && is_blank) ? 1 : 0);
    enc_t[b] = et;
    n_em[b] = oa ? (is_blank ? 0 : n_em[b] + 1) : n_em[b];
    int na = (oa && (et < enc_len[b])) ? 1 : 0;
    active[b] = na;
    int tok_out = (na || is_blank) ? tok : VV;
    out[(long)b * ITERS + step] = (float)tok_out;   // tokens [B, ITERS]
    out[(long)BB * ITERS + b]   = scores[b];        // final scores (last step's write persists)
    tok_sel[b] = tok > VV - 1 ? VV - 1 : tok;       // clip for embedding gather
  }
}

// ---- gates = E[tok] @ Wi + h @ Wh + bi + bh ----
__global__ void __launch_bounds__(32, 1)
gates_kernel(const bf16_t* __restrict__ Ebf, const int* __restrict__ tok_sel,
             const bf16_t* __restrict__ hbf, const bf16_t* __restrict__ WiT,
             const bf16_t* __restrict__ WhT, const float* __restrict__ bi,
             const float* __restrict__ bh, float* __restrict__ G) {
  int tile = blockIdx.x;                 // 8 * (2560/64=40)
  int m0 = (tile / 40) * 16, n0 = (tile % 40) * 64;
  v8f acc[4] = {};
  wmma_accum4<EMBD>(Ebf, EMBD, WiT, EMBD, m0, n0, tok_sel, acc);
  wmma_accum4<PREDD>(hbf, PREDD, WhT, PREDD, m0, n0, nullptr, acc);
  int lane = threadIdx.x & 31, r = lane & 15, half = lane >> 4;
#pragma unroll
  for (int nt = 0; nt < 4; ++nt) {
    int n = n0 + nt * 16 + r;
    float bsum = bi[n] + bh[n];
#pragma unroll
    for (int i = 0; i < 8; ++i)
      G[(long)(m0 + half * 8 + i) * GATES + n] = acc[nt][i] + bsum;
  }
}

// ---- masked LSTM state commit ----
__global__ void lstm_update_kernel(const float* __restrict__ G, const int* __restrict__ upd,
                                   float* __restrict__ h, float* __restrict__ c) {
  int b = blockIdx.x;
  int j = blockIdx.y * 256 + threadIdx.x;
  if (j >= PREDD || !upd[b]) return;
  const float* g = G + (long)b * GATES;
  float ig = sigmf(g[j]), fg = sigmf(g[640 + j]);
  float gg = tanhf(g[1280 + j]), og = sigmf(g[1920 + j]);
  long idx = (long)b * PREDD + j;
  float c2 = fg * c[idx] + ig * gg;
  c[idx] = c2;
  h[idx] = tanhf(c2) * og;
}

extern "C" void kernel_launch(void* const* d_in, const int* in_sizes, int n_in,
                              void* d_out, int out_size, void* d_ws, size_t ws_size,
                              hipStream_t stream) {
  (void)in_sizes; (void)n_in; (void)out_size; (void)ws_size;
  const float* enc_out = (const float*)d_in[0];
  const int*   enc_len = (const int*)d_in[1];
  const float* E       = (const float*)d_in[3];
  const float* Wi      = (const float*)d_in[4];
  const float* Wh      = (const float*)d_in[5];
  const float* bi      = (const float*)d_in[6];
  const float* bh      = (const float*)d_in[7];
  const float* W_enc   = (const float*)d_in[8];
  const float* W_pred  = (const float*)d_in[9];
  const float* b_joint = (const float*)d_in[10];
  const float* W_out   = (const float*)d_in[11];
  float* out = (float*)d_out;

  char* ws = (char*)d_ws;
  size_t off = 0;
  auto alloc = [&](size_t bytes) -> char* {
    char* p = ws + off; off = (off + bytes + 255) & ~(size_t)255; return p;
  };
  bf16_t* WencT  = (bf16_t*)alloc((size_t)JOINTD * ENCD * 2);
  bf16_t* WpredT = (bf16_t*)alloc((size_t)JOINTD * PREDD * 2);
  bf16_t* WoutT  = (bf16_t*)alloc((size_t)VV * JOINTD * 2);
  bf16_t* WiT    = (bf16_t*)alloc((size_t)GATES * EMBD * 2);
  bf16_t* WhT    = (bf16_t*)alloc((size_t)GATES * PREDD * 2);
  bf16_t* Ebf    = (bf16_t*)alloc((size_t)VV * EMBD * 2);
  bf16_t* enc_bf = (bf16_t*)alloc((size_t)BB * ENCD * 2);
  bf16_t* h_bf   = (bf16_t*)alloc((size_t)BB * PREDD * 2);
  bf16_t* Jbf    = (bf16_t*)alloc((size_t)BB * JOINTD * 2);
  float*  logits = (float*)alloc((size_t)BB * VV * 4);
  float*  G      = (float*)alloc((size_t)BB * GATES * 4);
  float*  h      = (float*)alloc((size_t)BB * PREDD * 4);
  float*  c      = (float*)alloc((size_t)BB * PREDD * 4);
  float*  g0     = (float*)alloc((size_t)GATES * 4);
  int*    enc_t  = (int*)alloc(BB * 4);
  int*    n_em   = (int*)alloc(BB * 4);
  float*  scores = (float*)alloc(BB * 4);
  int*    active = (int*)alloc(BB * 4);
  int*    tok_sel= (int*)alloc(BB * 4);
  int*    upd    = (int*)alloc(BB * 4);

  // one-time: bf16 weight prep (transposed for B-operand contiguity)
  transpose_cast_kernel<<<(ENCD * JOINTD) / 256, 256, 0, stream>>>(W_enc, WencT, ENCD, JOINTD);
  transpose_cast_kernel<<<(PREDD * JOINTD) / 256, 256, 0, stream>>>(W_pred, WpredT, PREDD, JOINTD);
  transpose_cast_kernel<<<(JOINTD * VV) / 256, 256, 0, stream>>>(W_out, WoutT, JOINTD, VV);
  transpose_cast_kernel<<<(EMBD * GATES) / 256, 256, 0, stream>>>(Wi, WiT, EMBD, GATES);
  transpose_cast_kernel<<<(PREDD * GATES) / 256, 256, 0, stream>>>(Wh, WhT, PREDD, GATES);
  cast_kernel<<<(VV * EMBD) / 256, 256, 0, stream>>>(E, Ebf, (long)VV * EMBD);

  // init predictor state from blank token, zero LSTM state
  init_gates_kernel<<<(GATES + 255) / 256, 256, 0, stream>>>(E, Wi, bi, bh, g0);
  init_state_kernel<<<dim3(BB, 3), 256, 0, stream>>>(g0, h, c, enc_t, n_em, scores, active);

  for (int step = 0; step < ITERS; ++step) {
    prep_kernel<<<dim3(BB, ENCD / 256), 256, 0, stream>>>(enc_out, enc_t, h, enc_bf, h_bf);
    joint_kernel<<<(BB / 16) * (JOINTD / 64), 32, 0, stream>>>(enc_bf, h_bf, WencT, WpredT,
                                                               b_joint, Jbf);
    logits_kernel<<<(BB / 16) * (VV / 64), 32, 0, stream>>>(Jbf, WoutT, logits);
    decide_kernel<<<BB, 256, 0, stream>>>(logits, enc_len, enc_t, n_em, scores, active,
                                          tok_sel, upd, out, step);
    gates_kernel<<<(BB / 16) * (GATES / 64), 32, 0, stream>>>(Ebf, tok_sel, h_bf, WiT, WhT,
                                                              bi, bh, G);
    lstm_update_kernel<<<dim3(BB, 3), 256, 0, stream>>>(G, upd, h, c);
  }
}
